// GCN_4758823764123
// MI455X (gfx1250) — matrix operand order repaired and checked by reading, not compiled, not run on hardware
//
#include <hip/hip_runtime.h>

#define H 128

typedef __attribute__((ext_vector_type(16))) __bf16 v16bf;
typedef __attribute__((ext_vector_type(8)))  float  v8f;
typedef __attribute__((ext_vector_type(4)))  float  v4f;

// ---------------- utility kernels ----------------

__global__ void fill_f32(float* p, long n, float v) {
  long i = (long)blockIdx.x * blockDim.x + threadIdx.x;
  long stride = (long)gridDim.x * blockDim.x;
  for (; i < n; i += stride) p[i] = v;
}

__global__ void deg_accum(const int* __restrict__ dst, int E, float* __restrict__ deg) {
  int e = blockIdx.x * blockDim.x + threadIdx.x;
  if (e < E) atomicAdd(&deg[dst[e]], 1.0f);
}

__global__ void dinv_inplace(float* deg, int N) {
  int i = blockIdx.x * blockDim.x + threadIdx.x;
  if (i < N) deg[i] = rsqrtf(deg[i]);  // deg >= 1 due to self-loop init
}

__global__ void coef_kernel(const int* __restrict__ src, const int* __restrict__ dst,
                            const float* __restrict__ dinv, int E, float* __restrict__ coef) {
  int e = blockIdx.x * blockDim.x + threadIdx.x;
  if (e < E) coef[e] = dinv[src[e]] * dinv[dst[e]];
}

// Pack fp32 W[K=128][N=128] (row-major) into bf16 WMMA B-operand lane layout:
// wp[((kc*8+nt)*32+lane)*16 + i] = bf16( W[kc*32 + 8*(lane>>4) + (i&7) + 16*(i>>3)][nt*16 + (lane&15)] )
__global__ void pack_w(const float* __restrict__ W, unsigned short* __restrict__ wp) {
  int tid = blockIdx.x * blockDim.x + threadIdx.x;
  if (tid >= H * H) return;
  int i    = tid & 15;
  int lane = (tid >> 4) & 31;
  int t    = tid >> 9;
  int nt   = t & 7;
  int kc   = t >> 3;
  int k = kc * 32 + 8 * (lane >> 4) + (i & 7) + 16 * (i >> 3);
  int n = nt * 16 + (lane & 15);
  unsigned int u = __float_as_uint(W[k * H + n]);
  unsigned int r = (u + 0x7FFFu + ((u >> 16) & 1u)) >> 16;  // RNE fp32 -> bf16
  wp[tid] = (unsigned short)r;
}

// ---------------- WMMA GEMM (optionally fused BN+ReLU on input) ----------------
// out[N x 128] = f(act)[N x 128] @ Wpacked, f(v) = relu(v*scale[k]+shift[k]) if scale!=null
__global__ void __launch_bounds__(256)
gemm_bn_relu(const float* __restrict__ act, const unsigned short* __restrict__ wp,
             float* __restrict__ out, const float* __restrict__ bnscale,
             const float* __restrict__ bnshift, int Nn) {
  const int lane  = threadIdx.x & 31;
  const int wave  = threadIdx.x >> 5;
  const int mtile = blockIdx.x * 8 + wave;
  const int nMtiles = (Nn + 15) >> 4;
  if (mtile >= nMtiles) return;            // wave-uniform exit: EXEC stays all-1s

  const int hi = lane >> 4;
  const int m  = mtile * 16 + (lane & 15); // same row for both lane halves (A layout)
  const bool mval = (m < Nn);
  const float* __restrict__ row = act + (size_t)m * H;

  v8f acc[8] = {};

#pragma unroll
  for (int kc = 0; kc < 4; ++kc) {
    const int b0 = kc * 32 + 8 * hi;       // k of halves i=0..7 ; i=8..15 at b0+16
    float f[16];
    if (mval) {
      v4f p0 = *(const v4f*)(row + b0);
      v4f p1 = *(const v4f*)(row + b0 + 4);
      v4f p2 = *(const v4f*)(row + b0 + 16);
      v4f p3 = *(const v4f*)(row + b0 + 20);
#pragma unroll
      for (int j = 0; j < 4; ++j) { f[j] = p0[j]; f[4+j] = p1[j]; f[8+j] = p2[j]; f[12+j] = p3[j]; }
    } else {
#pragma unroll
      for (int j = 0; j < 16; ++j) f[j] = 0.0f;
    }
    if (bnscale) {  // fused BN + ReLU of previous layer (kernel-arg uniform branch)
#pragma unroll
      for (int j = 0; j < 8; ++j) {
        int k0 = b0 + j, k1 = b0 + 16 + j;
        f[j]     = fmaxf(f[j]     * bnscale[k0] + bnshift[k0], 0.0f);
        f[8 + j] = fmaxf(f[8 + j] * bnscale[k1] + bnshift[k1], 0.0f);
      }
    }
    v16bf a;
#pragma unroll
    for (int j = 0; j < 16; ++j) a[j] = (__bf16)f[j];

#pragma unroll
    for (int nt = 0; nt < 8; ++nt) {
      v16bf b = *(const v16bf*)(wp + (size_t)((kc * 8 + nt) * 32 + lane) * 16);
      acc[nt] = __builtin_amdgcn_wmma_f32_16x16x32_bf16(
          false, a, false, b, (short)0, acc[nt], false, false);
    }
  }

  // C/D layout: VGPR r, lanes 0-15 -> M=r ; lanes 16-31 -> M=8+r ; N = lane&15
#pragma unroll
  for (int nt = 0; nt < 8; ++nt) {
    int col = nt * 16 + (lane & 15);
#pragma unroll
    for (int r = 0; r < 8; ++r) {
      int rowi = mtile * 16 + hi * 8 + r;
      if (rowi < Nn) out[(size_t)rowi * H + col] = acc[nt][r];
    }
  }
}

// ---------------- edge scatter (message passing), one wave per edge ----------------
__global__ void __launch_bounds__(256)
scatter_edges(const float* __restrict__ T, const float* __restrict__ coef,
              const int* __restrict__ src, const int* __restrict__ dst,
              int E, float* __restrict__ agg) {
  int lane = threadIdx.x & 31;
  int e = blockIdx.x * (blockDim.x >> 5) + (threadIdx.x >> 5);
  if (e >= E) return;
  int s = src[e], d = dst[e];
  float c = coef[e];
  v4f v = *(const v4f*)(T + (size_t)s * H + lane * 4);
  float* o = agg + (size_t)d * H + lane * 4;
  atomicAdd(o + 0, v[0] * c);
  atomicAdd(o + 1, v[1] * c);
  atomicAdd(o + 2, v[2] * c);
  atomicAdd(o + 3, v[3] * c);
}

// self-loop contribution + bias: agg[i][f] += T[i][f]*dinv[i]^2 + b[f]
__global__ void node_selfloop_bias(float* __restrict__ agg, const float* __restrict__ T,
                                   const float* __restrict__ dinv, const float* __restrict__ b,
                                   long total) {
  long idx = (long)blockIdx.x * blockDim.x + threadIdx.x;
  if (idx >= total) return;
  long i = idx >> 7;
  int  f = (int)(idx & 127);
  float di = dinv[i];
  agg[idx] += T[idx] * di * di + b[f];
}

// per-feature sum / sumsq partials via atomics; stats[0..127]=sum, [128..255]=sumsq
__global__ void bn_stats(const float* __restrict__ x, int Nn, float* __restrict__ stats) {
  int f   = threadIdx.x & 127;
  int sub = threadIdx.x >> 7;  // 0..1
  float s = 0.0f, q = 0.0f;
  for (int r = blockIdx.x * 2 + sub; r < Nn; r += gridDim.x * 2) {
    float v = x[(size_t)r * H + f];
    s += v; q += v * v;
  }
  atomicAdd(&stats[f], s);
  atomicAdd(&stats[H + f], q);
}

__global__ void bn_finalize(const float* __restrict__ stats, const float* __restrict__ g,
                            const float* __restrict__ be, int Nn,
                            float* __restrict__ scale, float* __restrict__ shift) {
  int f = threadIdx.x;
  if (f >= H) return;
  float inv = 1.0f / (float)Nn;
  float mean = stats[f] * inv;
  float var  = fmaxf(stats[H + f] * inv - mean * mean, 0.0f);
  float sc = g[f] * rsqrtf(var + 1e-5f);
  scale[f] = sc;
  shift[f] = be[f] - mean * sc;
}

// apply BN3+ReLU and accumulate per-graph sums / counts
__global__ void pool_accum(const float* __restrict__ agg, const float* __restrict__ scale,
                           const float* __restrict__ shift, const int* __restrict__ batch,
                           float* __restrict__ psum, float* __restrict__ pcnt, long total) {
  long idx = (long)blockIdx.x * blockDim.x + threadIdx.x;
  if (idx >= total) return;
  long i = idx >> 7;
  int  f = (int)(idx & 127);
  float v = fmaxf(agg[idx] * scale[f] + shift[f], 0.0f);
  int g = batch[i];
  atomicAdd(&psum[(size_t)g * H + f], v);
  if (f == 0) atomicAdd(&pcnt[g], 1.0f);
}

__global__ void readout(const float* __restrict__ psum, const float* __restrict__ pcnt,
                        const float* __restrict__ Wl, const float* __restrict__ bl,
                        float* __restrict__ out, int G) {
  int g = blockIdx.x * blockDim.x + threadIdx.x;
  if (g >= G) return;
  float inv = 1.0f / fmaxf(pcnt[g], 1.0f);
  float acc = bl[0];
  for (int f = 0; f < H; ++f) acc += psum[(size_t)g * H + f] * inv * Wl[f];
  out[g] = acc;
}

// ---------------- launcher ----------------

extern "C" void kernel_launch(void* const* d_in, const int* in_sizes, int n_in,
                              void* d_out, int out_size, void* d_ws, size_t ws_size,
                              hipStream_t stream) {
  (void)n_in; (void)ws_size;
  const float* x     = (const float*)d_in[0];
  const int*   ein   = (const int*)d_in[1];
  const int*   batch = (const int*)d_in[2];
  const float* W1 = (const float*)d_in[3];
  const float* b1 = (const float*)d_in[4];
  const float* g1 = (const float*)d_in[5];
  const float* be1= (const float*)d_in[6];
  const float* W2 = (const float*)d_in[7];
  const float* b2 = (const float*)d_in[8];
  const float* g2 = (const float*)d_in[9];
  const float* be2= (const float*)d_in[10];
  const float* W3 = (const float*)d_in[11];
  const float* b3 = (const float*)d_in[12];
  const float* g3 = (const float*)d_in[13];
  const float* be3= (const float*)d_in[14];
  const float* Wl = (const float*)d_in[15];
  const float* bl = (const float*)d_in[16];
  float* out = (float*)d_out;

  const int Nn = in_sizes[0] / H;   // 100000
  const int E  = in_sizes[1] / 2;   // 1600000
  const int G  = out_size;          // 256
  const int*   src = ein;
  const int*   dst = ein + E;

  // workspace carve (256B aligned slices)
  char* p = (char*)d_ws;
  auto carve = [&](size_t bytes) { char* r = p; p += (bytes + 255) & ~(size_t)255; return r; };
  float* deg   = (float*)carve(sizeof(float) * (size_t)Nn);         // -> dinv in place
  float* coef  = (float*)carve(sizeof(float) * (size_t)E);
  float* bufA  = (float*)carve(sizeof(float) * (size_t)Nn * H);
  float* bufB  = (float*)carve(sizeof(float) * (size_t)Nn * H);
  unsigned short* wp = (unsigned short*)carve(sizeof(unsigned short) * 3 * H * H);
  float* stats = (float*)carve(sizeof(float) * 2 * H);
  float* scl   = (float*)carve(sizeof(float) * 3 * H);
  float* sft   = (float*)carve(sizeof(float) * 3 * H);
  float* psum  = (float*)carve(sizeof(float) * (size_t)G * H);
  float* pcnt  = (float*)carve(sizeof(float) * (size_t)G);

  const int TB = 256;
  const long tot = (long)Nn * H;
  const int nMtiles = (Nn + 15) / 16;
  const int gemmBlocks = (nMtiles + 7) / 8;
  const int edgeBlocks = (E + 7) / 8;           // 8 waves (edges) per block
  const int totBlocks  = (int)((tot + TB - 1) / TB);

  // --- normalization precompute ---
  fill_f32<<<512, TB, 0, stream>>>(deg, Nn, 1.0f);              // self-loop degree
  deg_accum<<<(E + TB - 1) / TB, TB, 0, stream>>>(dst, E, deg);
  dinv_inplace<<<(Nn + TB - 1) / TB, TB, 0, stream>>>(deg, Nn); // deg -> deg^-1/2
  coef_kernel<<<(E + TB - 1) / TB, TB, 0, stream>>>(src, dst, deg, E, coef);
  pack_w<<<(H * H + TB - 1) / TB, TB, 0, stream>>>(W1, wp);
  pack_w<<<(H * H + TB - 1) / TB, TB, 0, stream>>>(W2, wp + H * H);
  pack_w<<<(H * H + TB - 1) / TB, TB, 0, stream>>>(W3, wp + 2 * H * H);

  // --- layer 1 ---
  gemm_bn_relu<<<gemmBlocks, 256, 0, stream>>>(x, wp, bufB, nullptr, nullptr, Nn);
  fill_f32<<<2048, TB, 0, stream>>>(bufA, tot, 0.0f);
  scatter_edges<<<edgeBlocks, 256, 0, stream>>>(bufB, coef, src, dst, E, bufA);
  node_selfloop_bias<<<totBlocks, TB, 0, stream>>>(bufA, bufB, deg, b1, tot);
  fill_f32<<<1, 256, 0, stream>>>(stats, 2 * H, 0.0f);
  bn_stats<<<512, 256, 0, stream>>>(bufA, Nn, stats);
  bn_finalize<<<1, H, 0, stream>>>(stats, g1, be1, Nn, scl, sft);

  // --- layer 2 (BN1+ReLU fused into GEMM input) ---
  gemm_bn_relu<<<gemmBlocks, 256, 0, stream>>>(bufA, wp + H * H, bufB, scl, sft, Nn);
  fill_f32<<<2048, TB, 0, stream>>>(bufA, tot, 0.0f);
  scatter_edges<<<edgeBlocks, 256, 0, stream>>>(bufB, coef, src, dst, E, bufA);
  node_selfloop_bias<<<totBlocks, TB, 0, stream>>>(bufA, bufB, deg, b2, tot);
  fill_f32<<<1, 256, 0, stream>>>(stats, 2 * H, 0.0f);
  bn_stats<<<512, 256, 0, stream>>>(bufA, Nn, stats);
  bn_finalize<<<1, H, 0, stream>>>(stats, g2, be2, Nn, scl + H, sft + H);

  // --- layer 3 ---
  gemm_bn_relu<<<gemmBlocks, 256, 0, stream>>>(bufA, wp + 2 * H * H, bufB, scl + H, sft + H, Nn);
  fill_f32<<<2048, TB, 0, stream>>>(bufA, tot, 0.0f);
  scatter_edges<<<edgeBlocks, 256, 0, stream>>>(bufB, coef, src, dst, E, bufA);
  node_selfloop_bias<<<totBlocks, TB, 0, stream>>>(bufA, bufB, deg, b3, tot);
  fill_f32<<<1, 256, 0, stream>>>(stats, 2 * H, 0.0f);
  bn_stats<<<512, 256, 0, stream>>>(bufA, Nn, stats);
  bn_finalize<<<1, H, 0, stream>>>(stats, g3, be3, Nn, scl + 2 * H, sft + 2 * H);

  // --- global mean pool + readout ---
  fill_f32<<<256, TB, 0, stream>>>(psum, (long)G * H, 0.0f);
  fill_f32<<<1, TB, 0, stream>>>(pcnt, G, 0.0f);
  pool_accum<<<totBlocks, TB, 0, stream>>>(bufA, scl + 2 * H, sft + 2 * H, batch, psum, pcnt, tot);
  readout<<<(G + 255) / 256, 256, 0, stream>>>(psum, pcnt, Wl, bl, out, G);
}
